// TimesformerLayer_31903017074699
// MI455X (gfx1250) — compile-verified, matrix operand order checked by reading, not run on hardware
//
#include <hip/hip_runtime.h>
#include <cstdint>
#include <cstddef>

// ---------------------------------------------------------------------------
// Types for CDNA5 WMMA (wave32, V_WMMA_F32_16X16X32_BF16)
// ---------------------------------------------------------------------------
typedef __attribute__((ext_vector_type(16))) __bf16 v16bf;
typedef __attribute__((ext_vector_type(8)))  float  v8f;

#define DEV static __device__ __forceinline__

DEV unsigned short f2bf(float f) {
    union { float f; uint32_t u; } x; x.f = f;
    uint32_t r = x.u + 0x7FFFu + ((x.u >> 16) & 1u);   // round-to-nearest-even
    return (unsigned short)(r >> 16);
}

DEV float gelu_exact(float x) {
    return 0.5f * x * (1.0f + erff(x * 0.70710678118654752f));
}

// low 32 bits of a generic "shared" pointer == wave-relative LDS byte offset
DEV uint32_t lds_offset(const void* p) {
    return (uint32_t)(uintptr_t)p;
}

// CDNA5 async global->LDS copy (16B), tracked by ASYNCcnt (ISA ch.10 / §15.18)
DEV void async_copy_b128(uint32_t lds_off, const void* gptr) {
    asm volatile("global_load_async_to_lds_b128 %0, %1, off"
                 :: "v"(lds_off), "v"((uint64_t)(uintptr_t)gptr)
                 : "memory");
}
DEV void wait_asynccnt0() {
    asm volatile("s_wait_asynccnt 0x0" ::: "memory");
}

// ---------------------------------------------------------------------------
// Weight transpose + fp32 -> bf16 : Wt[n*K + k] = bf16(W[k*N + n])
// ---------------------------------------------------------------------------
__global__ void wT_to_bf16(const float* __restrict__ W, unsigned short* __restrict__ Wt,
                           int K, int N) {
    size_t total = (size_t)K * (size_t)N;
    for (size_t i = (size_t)blockIdx.x * 256u + threadIdx.x; i < total;
         i += (size_t)gridDim.x * 256u) {
        size_t n = i / (size_t)K, k = i % (size_t)K;
        Wt[i] = f2bf(W[k * (size_t)N + n]);
    }
}

// ---------------------------------------------------------------------------
// Tiled bf16 WMMA GEMM: D[M,N] = A[M,K] * Bt[N,K]^T  (+bias, epilogue)
//   EPI 0: fp32 out = acc + bias
//   EPI 1: fp32 out = acc + bias + res[row*N+col]
//   EPI 2: bf16 out = gelu(acc + bias)
//   EPI 3: bf16 out = acc + bias
// BM=BN=128, BK=32. 256 threads = 8 waves; wave grid 4(M) x 2(N); each wave
// computes 32x64 via 2x4 WMMA tiles.  Tiles are fed with double-buffered
// GLOBAL_LOAD_ASYNC_TO_LDS_B128 (ASYNCcnt), overlapping DMA with WMMA.
// N, K must be multiples of 128 / 32 (true for 768/2304/3072); M is handled
// by clamping A rows (affects only unstored C rows) + store-side guard.
// ---------------------------------------------------------------------------
template <int EPI>
__global__ __launch_bounds__(256)
void gemm_bf16_wmma(const unsigned short* __restrict__ A,
                    const unsigned short* __restrict__ Bt,
                    const float* __restrict__ bias,
                    const float* __restrict__ res,
                    void* __restrict__ dst,
                    int M, int N, int K) {
    __shared__ alignas(32) unsigned short As[2][128 * 32];
    __shared__ alignas(32) unsigned short Bs[2][128 * 32];

    const int tid  = threadIdx.x;
    const int lane = tid & 31;
    const int wid  = tid >> 5;
    const int mBase = blockIdx.y * 128;
    const int nBase = blockIdx.x * 128;
    const int wm = (wid >> 1) * 32;     // 0,32,64,96
    const int wn = (wid & 1) * 64;      // 0,64
    const int laneM  = lane & 15;
    const int laneKh = (lane >> 4) * 16;

    // per-thread tile-chunk geometry: 2 chunks of 16B per tile per operand
    int crow[2], ccol[2], arow[2];
    #pragma unroll
    for (int it = 0; it < 2; ++it) {
        int c = tid + it * 256;        // 0..511
        crow[it] = c >> 2;             // 4 chunks of 8 elems per 32-elem row
        ccol[it] = (c & 3) * 8;
        int ga = mBase + crow[it];
        arow[it] = (ga < M) ? ga : (M - 1);   // clamp: tail rows never stored
    }
    const uint32_t asBase = lds_offset(&As[0][0]);
    const uint32_t bsBase = lds_offset(&Bs[0][0]);
    const uint32_t bufStride = 128 * 32 * 2;  // bytes per buffer

    auto issue_tile = [&](int k0, int buf) {
        #pragma unroll
        for (int it = 0; it < 2; ++it) {
            uint32_t loff = (uint32_t)(crow[it] * 32 + ccol[it]) * 2 + buf * bufStride;
            async_copy_b128(asBase + loff, A  + (size_t)arow[it] * K + k0 + ccol[it]);
            async_copy_b128(bsBase + loff, Bt + (size_t)(nBase + crow[it]) * K + k0 + ccol[it]);
        }
    };

    v8f acc[2][4] = {};
    const int nk = K / 32;

    issue_tile(0, 0);
    wait_asynccnt0();
    __syncthreads();

    for (int t = 0; t < nk; ++t) {
        const int cur = t & 1;
        if (t + 1 < nk) issue_tile((t + 1) * 32, cur ^ 1);

        // fragments: lanes 0-15 hold K=0..15, lanes 16-31 hold K=16..31
        v16bf af[2], bfr[4];
        #pragma unroll
        for (int i = 0; i < 2; ++i)
            af[i] = *reinterpret_cast<const v16bf*>(&As[cur][(wm + i * 16 + laneM) * 32 + laneKh]);
        #pragma unroll
        for (int j = 0; j < 4; ++j)
            bfr[j] = *reinterpret_cast<const v16bf*>(&Bs[cur][(wn + j * 16 + laneM) * 32 + laneKh]);

        #pragma unroll
        for (int i = 0; i < 2; ++i)
            #pragma unroll
            for (int j = 0; j < 4; ++j)
                acc[i][j] = __builtin_amdgcn_wmma_f32_16x16x32_bf16(
                    false, af[i], false, bfr[j], (short)0, acc[i][j], false, false);

        wait_asynccnt0();   // next tile's async batch complete before reuse
        __syncthreads();
    }

    // C/D layout: lanes 0-15: M=r, N=lane ; lanes 16-31: M=8+r, N=lane-16
    const int mo = (lane >> 4) * 8;
    const int no = lane & 15;
    float*          of = (float*)dst;
    unsigned short* oh = (unsigned short*)dst;

    #pragma unroll
    for (int i = 0; i < 2; ++i)
        #pragma unroll
        for (int j = 0; j < 4; ++j)
            #pragma unroll
            for (int r = 0; r < 8; ++r) {
                int grow = mBase + wm + i * 16 + mo + r;
                int gcol = nBase + wn + j * 16 + no;
                if (grow >= M) continue;
                float x = acc[i][j][r] + bias[gcol];
                size_t o = (size_t)grow * N + gcol;
                if constexpr (EPI == 0) { of[o] = x; }
                else if constexpr (EPI == 1) { of[o] = x + res[o]; }
                else if constexpr (EPI == 2) { oh[o] = f2bf(gelu_exact(x)); }
                else { oh[o] = f2bf(x); }
            }
}

// ---------------------------------------------------------------------------
// LayerNorm over C=768, fp32 in -> bf16 out (feeds GEMM A operand)
// ---------------------------------------------------------------------------
__global__ __launch_bounds__(256)
void layernorm_to_bf16(const float* __restrict__ X, const float* __restrict__ g,
                       const float* __restrict__ b, unsigned short* __restrict__ Y,
                       int rows) {
    const int row = blockIdx.x;
    if (row >= rows) return;
    const float* x = X + (size_t)row * 768;
    float v[3], s1 = 0.f, s2 = 0.f;
    #pragma unroll
    for (int k = 0; k < 3; ++k) {
        float t = x[threadIdx.x + k * 256];
        v[k] = t; s1 += t; s2 += t * t;
    }
    #pragma unroll
    for (int off = 16; off; off >>= 1) {
        s1 += __shfl_xor(s1, off, 32);
        s2 += __shfl_xor(s2, off, 32);
    }
    __shared__ float ws1[8], ws2[8], stat[2];
    int lane = threadIdx.x & 31, w = threadIdx.x >> 5;
    if (lane == 0) { ws1[w] = s1; ws2[w] = s2; }
    __syncthreads();
    if (threadIdx.x == 0) {
        float t1 = 0.f, t2 = 0.f;
        for (int i = 0; i < 8; ++i) { t1 += ws1[i]; t2 += ws2[i]; }
        float mean = t1 * (1.0f / 768.0f);
        float var  = t2 * (1.0f / 768.0f) - mean * mean;
        stat[0] = mean; stat[1] = rsqrtf(var + 1e-6f);
    }
    __syncthreads();
    float mean = stat[0], rstd = stat[1];
    #pragma unroll
    for (int k = 0; k < 3; ++k) {
        int c = threadIdx.x + k * 256;
        Y[(size_t)row * 768 + c] = f2bf((v[k] - mean) * rstd * g[c] + b[c]);
    }
}

// ---------------------------------------------------------------------------
// Temporal attention: 1568 batches x 12 heads, T=8, hd=64.  QKV fp32 rows
// ordered (b,hw,t), cols [q|k|v] each 12*64.  Output bf16, same row order.
// ---------------------------------------------------------------------------
__global__ __launch_bounds__(64)
void temporal_attn(const float* __restrict__ QKV, unsigned short* __restrict__ O) {
    const int blk = blockIdx.x;
    const int p = blk / 12, h = blk % 12;
    const int tid = threadIdx.x;
    __shared__ float Ks[8][64], Vs[8][64], Sc[8][8];
    const size_t base = (size_t)p * 8 * 2304 + h * 64;

    for (int idx = tid; idx < 512; idx += 64) {
        int t = idx >> 6, d = idx & 63;
        Ks[t][d] = QKV[base + (size_t)t * 2304 + 768 + d];
        Vs[t][d] = QKV[base + (size_t)t * 2304 + 1536 + d];
    }
    __syncthreads();
    {   // 64 threads == 8x8 score matrix
        int tq = tid >> 3, tk = tid & 7;
        float s = 0.f;
        for (int d = 0; d < 64; ++d)
            s += QKV[base + (size_t)tq * 2304 + d] * Ks[tk][d];
        Sc[tq][tk] = s * 0.125f;   // hd^-0.5
    }
    __syncthreads();
    for (int idx = tid; idx < 512; idx += 64) {
        int tq = idx >> 6, d = idx & 63;
        float mx = -1e30f;
        for (int tk = 0; tk < 8; ++tk) mx = fmaxf(mx, Sc[tq][tk]);
        float den = 0.f, o = 0.f;
        for (int tk = 0; tk < 8; ++tk) {
            float e = __expf(Sc[tq][tk] - mx);
            den += e; o += e * Vs[tk][d];
        }
        O[((size_t)p * 8 + tq) * 768 + h * 64 + d] = f2bf(o / den);
    }
}

// ---------------------------------------------------------------------------
// Spatial attention: 64 batches x 12 heads, S=197, hd=64. Flash-style online
// softmax, 32-query x 32-key tiles, 28KB LDS.
// ---------------------------------------------------------------------------
__global__ __launch_bounds__(256)
void spatial_attn(const float* __restrict__ QKV, unsigned short* __restrict__ O) {
    const int qt = blockIdx.x, h = blockIdx.y, bb = blockIdx.z;
    const int tid = threadIdx.x;
    __shared__ float Qs[32][64], Ks[32][64], Vs[32][64], Sc[32][32];
    const size_t rowBase = (size_t)bb * 197;

    for (int idx = tid; idx < 2048; idx += 256) {
        int q = idx >> 6, d = idx & 63;
        int qg = qt * 32 + q;
        Qs[q][d] = (qg < 197) ? QKV[(rowBase + qg) * 2304 + h * 64 + d] : 0.0f;
    }

    const int row = tid >> 3, sub = tid & 7;   // 8 lanes per query row
    float mrun = -1e30f, lrun = 0.0f;
    float ctx[8] = {0, 0, 0, 0, 0, 0, 0, 0};

    for (int j0 = 0; j0 < 197; j0 += 32) {
        __syncthreads();
        for (int idx = tid; idx < 2048; idx += 256) {
            int j = idx >> 6, d = idx & 63;
            int jg = j0 + j;
            float kv = (jg < 197) ? QKV[(rowBase + jg) * 2304 + 768 + h * 64 + d] : 0.0f;
            float vv = (jg < 197) ? QKV[(rowBase + jg) * 2304 + 1536 + h * 64 + d] : 0.0f;
            Ks[j][d] = kv; Vs[j][d] = vv;
        }
        __syncthreads();
        for (int s = tid; s < 1024; s += 256) {
            int q = s >> 5, j = s & 31;
            float a = 0.f;
            for (int d = 0; d < 64; ++d) a += Qs[q][d] * Ks[j][d];
            Sc[q][j] = (j0 + j < 197) ? a * 0.125f : -1e30f;
        }
        __syncthreads();

        float tmax = -1e30f;
        for (int j = sub; j < 32; j += 8) tmax = fmaxf(tmax, Sc[row][j]);
        #pragma unroll
        for (int off = 4; off; off >>= 1) tmax = fmaxf(tmax, __shfl_xor(tmax, off, 8));
        float mnew = fmaxf(mrun, tmax);
        float corr = __expf(mrun - mnew);
        float lp = 0.f;
        for (int j = sub; j < 32; j += 8) lp += __expf(Sc[row][j] - mnew);
        #pragma unroll
        for (int off = 4; off; off >>= 1) lp += __shfl_xor(lp, off, 8);
        lrun = lrun * corr + lp;
        mrun = mnew;
        #pragma unroll
        for (int dd = 0; dd < 8; ++dd) ctx[dd] *= corr;
        for (int j = 0; j < 32; ++j) {
            float pj = __expf(Sc[row][j] - mnew);
            #pragma unroll
            for (int dd = 0; dd < 8; ++dd) ctx[dd] += pj * Vs[j][sub * 8 + dd];
        }
    }

    int qg = qt * 32 + row;
    if (qg < 197) {
        float inv = 1.0f / lrun;
        #pragma unroll
        for (int dd = 0; dd < 8; ++dd)
            O[(rowBase + qg) * 768 + h * 64 + sub * 8 + dd] = f2bf(ctx[dd] * inv);
    }
}

// ---------------------------------------------------------------------------
// Glue kernels (reshapes / residuals / cls handling), grid-stride elementwise
// ---------------------------------------------------------------------------
__global__ void copy_nocls(const float* __restrict__ hid, float* __restrict__ X) {
    // X[b*1568+r] = hid[b*1569+1+r]   (rows of C=768)
    size_t total = (size_t)12544 * 768;
    for (size_t i = (size_t)blockIdx.x * 256u + threadIdx.x; i < total;
         i += (size_t)gridDim.x * 256u) {
        size_t r = i / 768, c = i % 768;
        size_t b = r / 1568, rr = r % 1568;
        X[i] = hid[(b * 1569 + 1 + rr) * 768 + c];
    }
}

__global__ void build_semb(const float* __restrict__ hid, const float* __restrict__ TEMB,
                           float* __restrict__ SEMB) {
    size_t total = (size_t)12608 * 768;   // (64, 197, 768)
    for (size_t i = (size_t)blockIdx.x * 256u + threadIdx.x; i < total;
         i += (size_t)gridDim.x * 256u) {
        size_t row = i / 768, c = i % 768;
        size_t bb = row / 197, s = row % 197;
        size_t b = bb >> 3, t = bb & 7;
        float v;
        if (s == 0) v = hid[(b * 1569) * 768 + c];
        else {
            size_t hw = s - 1;
            v = TEMB[(b * 1568 + hw * 8 + t) * 768 + c];
        }
        SEMB[i] = v;
    }
}

__global__ void cls_mean(const float* __restrict__ SOUT, float* __restrict__ CLS) {
    size_t total = (size_t)8 * 768;
    for (size_t i = (size_t)blockIdx.x * 256u + threadIdx.x; i < total;
         i += (size_t)gridDim.x * 256u) {
        size_t b = i / 768, c = i % 768;
        float s = 0.f;
        for (int t = 0; t < 8; ++t)
            s += SOUT[((b * 8 + t) * 197) * 768 + c];
        CLS[i] = s * 0.125f;
    }
}

__global__ void build_hs(const float* __restrict__ hid, const float* __restrict__ TEMB,
                         const float* __restrict__ SOUT, const float* __restrict__ CLS,
                         float* __restrict__ HS) {
    size_t total = (size_t)12552 * 768;   // (8, 1569, 768)
    for (size_t i = (size_t)blockIdx.x * 256u + threadIdx.x; i < total;
         i += (size_t)gridDim.x * 256u) {
        size_t row = i / 768, c = i % 768;
        size_t b = row / 1569, s = row % 1569;
        float v;
        if (s == 0) v = hid[row * 768 + c] + CLS[b * 768 + c];
        else {
            size_t r = s - 1;
            size_t hw = r >> 3, t = r & 7;
            v = TEMB[(b * 1568 + r) * 768 + c] +
                SOUT[((b * 8 + t) * 197 + 1 + hw) * 768 + c];
        }
        HS[i] = v;
    }
}

// ---------------------------------------------------------------------------
// Launch
// ---------------------------------------------------------------------------
extern "C" void kernel_launch(void* const* d_in, const int* in_sizes, int n_in,
                              void* d_out, int out_size, void* d_ws, size_t ws_size,
                              hipStream_t stream) {
    (void)in_sizes; (void)n_in; (void)out_size; (void)ws_size;

    const float* hid     = (const float*)d_in[0];
    const float* ln_t_g  = (const float*)d_in[1];
    const float* ln_t_b  = (const float*)d_in[2];
    const float* t_qkv_w = (const float*)d_in[3];
    const float* t_qkv_b = (const float*)d_in[4];
    const float* t_prj_w = (const float*)d_in[5];
    const float* t_prj_b = (const float*)d_in[6];
    const float* t_dns_w = (const float*)d_in[7];
    const float* t_dns_b = (const float*)d_in[8];
    const float* ln1_g   = (const float*)d_in[9];
    const float* ln1_b   = (const float*)d_in[10];
    const float* s_qkv_w = (const float*)d_in[11];
    const float* s_qkv_b = (const float*)d_in[12];
    const float* s_prj_w = (const float*)d_in[13];
    const float* s_prj_b = (const float*)d_in[14];
    const float* ln2_g   = (const float*)d_in[15];
    const float* ln2_b   = (const float*)d_in[16];
    const float* fc1_w   = (const float*)d_in[17];
    const float* fc1_b   = (const float*)d_in[18];
    const float* fc2_w   = (const float*)d_in[19];
    const float* fc2_b   = (const float*)d_in[20];
    float* out = (float*)d_out;

    const int C = 768, NQ = 2304, I = 3072;
    const int Mt = 12544, Ms = 12608, Mh = 12552;

    // workspace bump allocator
    size_t off = 0;
    auto alloc = [&](size_t bytes) -> void* {
        size_t cur = (off + 255u) & ~(size_t)255u;
        off = cur + bytes;
        return (void*)((char*)d_ws + cur);
    };
    unsigned short* WqT_t = (unsigned short*)alloc((size_t)NQ * C * 2);
    unsigned short* WpT_t = (unsigned short*)alloc((size_t)C * C * 2);
    unsigned short* WdT   = (unsigned short*)alloc((size_t)C * C * 2);
    unsigned short* WqT_s = (unsigned short*)alloc((size_t)NQ * C * 2);
    unsigned short* WpT_s = (unsigned short*)alloc((size_t)C * C * 2);
    unsigned short* Wf1T  = (unsigned short*)alloc((size_t)I * C * 2);
    unsigned short* Wf2T  = (unsigned short*)alloc((size_t)C * I * 2);
    float* XNC  = (float*)alloc((size_t)Mt * C * 4);
    float* QKV  = (float*)alloc((size_t)Ms * NQ * 4);
    float* TEMB = (float*)alloc((size_t)Mt * C * 4);
    float* SEMB = (float*)alloc((size_t)Ms * C * 4);
    float* SOUT = (float*)alloc((size_t)Ms * C * 4);
    float* CLS  = (float*)alloc((size_t)8 * C * 4);
    float* HS   = (float*)alloc((size_t)Mh * C * 4);
    unsigned short* Aa = (unsigned short*)alloc((size_t)Ms * C * 2);
    unsigned short* Ab = (unsigned short*)alloc((size_t)Mt * C * 2);
    unsigned short* A6 = (unsigned short*)alloc((size_t)Mh * I * 2);

    const int EW = 4096;  // grid-stride elementwise blocks

    // --- weight prep (bf16, N x K) ---
    wT_to_bf16<<<EW, 256, 0, stream>>>(t_qkv_w, WqT_t, C, NQ);
    wT_to_bf16<<<EW, 256, 0, stream>>>(t_prj_w, WpT_t, C, C);
    wT_to_bf16<<<EW, 256, 0, stream>>>(t_dns_w, WdT,   C, C);
    wT_to_bf16<<<EW, 256, 0, stream>>>(s_qkv_w, WqT_s, C, NQ);
    wT_to_bf16<<<EW, 256, 0, stream>>>(s_prj_w, WpT_s, C, C);
    wT_to_bf16<<<EW, 256, 0, stream>>>(fc1_w,   Wf1T,  C, I);
    wT_to_bf16<<<EW, 256, 0, stream>>>(fc2_w,   Wf2T,  I, C);

    auto gemmGrid = [](int M, int N) { return dim3((unsigned)(N / 128), (unsigned)((M + 127) / 128)); };

    // --- temporal stage ---
    copy_nocls<<<EW, 256, 0, stream>>>(hid, XNC);
    layernorm_to_bf16<<<Mt, 256, 0, stream>>>(XNC, ln_t_g, ln_t_b, Aa, Mt);
    gemm_bf16_wmma<0><<<gemmGrid(Mt, NQ), 256, 0, stream>>>(Aa, WqT_t, t_qkv_b, nullptr, QKV, Mt, NQ, C);
    temporal_attn<<<1568 * 12, 64, 0, stream>>>(QKV, Aa);
    gemm_bf16_wmma<3><<<gemmGrid(Mt, C), 256, 0, stream>>>(Aa, WpT_t, t_prj_b, nullptr, Ab, Mt, C, C);
    gemm_bf16_wmma<1><<<gemmGrid(Mt, C), 256, 0, stream>>>(Ab, WdT, t_dns_b, XNC, TEMB, Mt, C, C);

    // --- spatial stage ---
    build_semb<<<EW, 256, 0, stream>>>(hid, TEMB, SEMB);
    layernorm_to_bf16<<<Ms, 256, 0, stream>>>(SEMB, ln1_g, ln1_b, Aa, Ms);
    gemm_bf16_wmma<0><<<gemmGrid(Ms, NQ), 256, 0, stream>>>(Aa, WqT_s, s_qkv_b, nullptr, QKV, Ms, NQ, C);
    spatial_attn<<<dim3(7, 12, 64), 256, 0, stream>>>(QKV, Aa);
    gemm_bf16_wmma<0><<<gemmGrid(Ms, C), 256, 0, stream>>>(Aa, WpT_s, s_prj_b, nullptr, SOUT, Ms, C, C);
    cls_mean<<<32, 256, 0, stream>>>(SOUT, CLS);
    build_hs<<<EW, 256, 0, stream>>>(hid, TEMB, SOUT, CLS, HS);

    // --- MLP ---
    layernorm_to_bf16<<<Mh, 256, 0, stream>>>(HS, ln2_g, ln2_b, Aa, Mh);
    gemm_bf16_wmma<2><<<gemmGrid(Mh, I), 256, 0, stream>>>(Aa, Wf1T, fc1_b, nullptr, A6, Mh, I, C);
    gemm_bf16_wmma<1><<<gemmGrid(Mh, C), 256, 0, stream>>>(A6, Wf2T, fc2_b, HS, out, Mh, C, I);
}
